// ScheduleDecoder_6889127543024
// MI455X (gfx1250) — matrix operand order repaired and checked by reading, not compiled, not run on hardware
//
#include <hip/hip_runtime.h>

// ---------------- problem constants (from reference setup_inputs) ------------
#define BATCH   1024
#define HID     512
#define NLAY    2
#define NGATE   2048          // 4*HID
#define EMBW    511           // HID-1
#define NOUT    17
#define NACT    16
#define TSTEPS  64

typedef __bf16        bf16x16 __attribute__((ext_vector_type(16)));
typedef float         v8f     __attribute__((ext_vector_type(8)));
typedef unsigned int  u32x4   __attribute__((ext_vector_type(4)));

union FragBF { u32x4 u[2]; bf16x16 v; };

__device__ __forceinline__ unsigned short f2bf(float f) {
    unsigned u = __float_as_uint(f);
    u += 0x7FFFu + ((u >> 16) & 1u);          // round-to-nearest-even
    return (unsigned short)(u >> 16);
}
__device__ __forceinline__ float bf2f(unsigned short u) {
    return __uint_as_float(((unsigned)u) << 16);
}
__device__ __forceinline__ v8f zero8() {
    v8f z;
#pragma unroll
    for (int i = 0; i < 8; ++i) z[i] = 0.0f;
    return z;
}

// A fragment: 16x32 (MxK) bf16, rows m0..m0+15, K chunk k0..k0+31.
// Lane l<16: K {0..15}; lane l>=16: K {16..31} (matches B fragment order).
__device__ __forceinline__ bf16x16
load_a_frag(const unsigned short* __restrict__ X, int ld, int m0, int k0, int lane) {
    const int row  = m0 + (lane & 15);
    const int koff = k0 + ((lane >> 4) << 4);
    const unsigned short* p = X + (size_t)row * ld + koff;
    FragBF f;
    f.u[0] = *(const u32x4*)(p);        // K koff..koff+7   -> VGPR0..3
    f.u[1] = *(const u32x4*)(p + 8);    // K koff+8..+15    -> VGPR4..7
    return f.v;
}

// B fragment: 32x16 (KxN) bf16 taken from W stored N-major (W[n][k], k contig).
// Lane l<16: col n0+l, K {0..15}; lane l>=16: col n0+l-16, K {16..31}.
__device__ __forceinline__ bf16x16
load_b_frag(const unsigned short* __restrict__ W, int ld, int n0, int k0, int lane) {
    const int col  = n0 + (lane & 15);
    const int koff = k0 + ((lane >> 4) << 4);
    const unsigned short* p = W + (size_t)col * ld + koff;
    FragBF f;
    f.u[0] = *(const u32x4*)(p);        // K koff..koff+7
    f.u[1] = *(const u32x4*)(p + 8);    // K koff+8..+15
    return f.v;
}

#define WMMA_BF16(A, B, C) \
    __builtin_amdgcn_wmma_f32_16x16x32_bf16(false, (A), false, (B), (short)0, (C), false, false)

// ---------------------------------------------------------------------------
// f32 -> bf16 (stored as ushort) elementwise conversion
__global__ __launch_bounds__(256)
void cvt_bf16_kernel(const float* __restrict__ src, unsigned short* __restrict__ dst, int n) {
    int i = blockIdx.x * 256 + threadIdx.x;
    if (i < n) dst[i] = f2bf(src[i]);
}

// build layer-0 input x[b][k]: emb_table[act[b]] (511) ++ dur
__global__ __launch_bounds__(256)
void embed_kernel(const float* __restrict__ emb, const int* __restrict__ act,
                  int first, unsigned short* __restrict__ X) {
    int idx = blockIdx.x * 256 + threadIdx.x;          // < BATCH*HID
    int b = idx >> 9, k = idx & 511;
    float v;
    if (k < EMBW) {
        int a = first ? 0 : act[b];
        v = emb[(size_t)a * EMBW + k];
    } else {
        v = first ? 0.0f : 1.0f;                        // duration input
    }
    X[idx] = f2bf(v);
}

// gates = X @ Wih^T + Hp @ Whh^T        (M=1024, N=2048, K=512, bf16 -> f32)
// block = 8 waves; wave tile 32x32 (2x2 WMMA tiles); block tile 64x128
__global__ __launch_bounds__(256)
void gates_kernel(const unsigned short* __restrict__ X,
                  const unsigned short* __restrict__ Hp,
                  const unsigned short* __restrict__ Wih,
                  const unsigned short* __restrict__ Whh,
                  float* __restrict__ G) {
    const int lane = threadIdx.x & 31;
    const int wave = threadIdx.x >> 5;
    const int m0 = blockIdx.x * 64  + (wave & 1) * 32;
    const int n0 = blockIdx.y * 128 + (wave >> 1) * 32;

    v8f acc[2][2];
#pragma unroll
    for (int i = 0; i < 2; ++i)
#pragma unroll
        for (int j = 0; j < 2; ++j) acc[i][j] = zero8();

#pragma unroll 2
    for (int k0 = 0; k0 < HID; k0 += 32) {
        // input-weight product
        bf16x16 a0 = load_a_frag(X, HID, m0,      k0, lane);
        bf16x16 a1 = load_a_frag(X, HID, m0 + 16, k0, lane);
        bf16x16 b0 = load_b_frag(Wih, HID, n0,      k0, lane);
        bf16x16 b1 = load_b_frag(Wih, HID, n0 + 16, k0, lane);
        acc[0][0] = WMMA_BF16(a0, b0, acc[0][0]);
        acc[0][1] = WMMA_BF16(a0, b1, acc[0][1]);
        acc[1][0] = WMMA_BF16(a1, b0, acc[1][0]);
        acc[1][1] = WMMA_BF16(a1, b1, acc[1][1]);
        // recurrent product into the same accumulators
        bf16x16 c0 = load_a_frag(Hp, HID, m0,      k0, lane);
        bf16x16 c1 = load_a_frag(Hp, HID, m0 + 16, k0, lane);
        bf16x16 d0 = load_b_frag(Whh, HID, n0,      k0, lane);
        bf16x16 d1 = load_b_frag(Whh, HID, n0 + 16, k0, lane);
        acc[0][0] = WMMA_BF16(c0, d0, acc[0][0]);
        acc[0][1] = WMMA_BF16(c0, d1, acc[0][1]);
        acc[1][0] = WMMA_BF16(c1, d0, acc[1][0]);
        acc[1][1] = WMMA_BF16(c1, d1, acc[1][1]);
    }

    // D layout: lane<16 -> N=lane, M=r ; lane>=16 -> N=lane-16, M=8+r
    const int cn = lane & 15;
    const int rm = (lane >> 4) * 8;
#pragma unroll
    for (int ti = 0; ti < 2; ++ti)
#pragma unroll
        for (int tj = 0; tj < 2; ++tj) {
            const int rowb = m0 + ti * 16 + rm;
            const int col  = n0 + tj * 16 + cn;
#pragma unroll
            for (int r = 0; r < 8; ++r)
                G[(size_t)(rowb + r) * NGATE + col] = acc[ti][tj][r];
        }
}

// LSTM pointwise: c' = sig(f)*c + sig(i)*tanh(g); h' = sig(o)*tanh(c')
__global__ __launch_bounds__(256)
void cell_kernel(const float* __restrict__ G, const float* __restrict__ bih,
                 const float* __restrict__ bhh, float* __restrict__ C,
                 unsigned short* __restrict__ Hout) {
    int idx = blockIdx.x * 256 + threadIdx.x;          // < BATCH*HID
    int b = idx >> 9, n = idx & 511;
    const float* g = G + (size_t)b * NGATE;
    float gi = g[n]            + bih[n]            + bhh[n];
    float gf = g[HID + n]      + bih[HID + n]      + bhh[HID + n];
    float gg = g[2 * HID + n]  + bih[2 * HID + n]  + bhh[2 * HID + n];
    float go = g[3 * HID + n]  + bih[3 * HID + n]  + bhh[3 * HID + n];
    float si = 1.0f / (1.0f + expf(-gi));
    float sf = 1.0f / (1.0f + expf(-gf));
    float so = 1.0f / (1.0f + expf(-go));
    float c  = sf * C[idx] + si * tanhf(gg);
    C[idx]   = c;
    Hout[idx] = f2bf(so * tanhf(c));
}

// pred = h1 @ fcW^T + fcb ; write logits to d_out[b][t][:]; argmax -> act[b]
__global__ __launch_bounds__(32)
void fc_kernel(const unsigned short* __restrict__ H1, const float* __restrict__ fcW,
               const float* __restrict__ fcb, float* __restrict__ out,
               int* __restrict__ act, int t) {
    int b = blockIdx.x, o = threadIdx.x;
    __shared__ float s[NOUT];
    if (o < NOUT) {
        const unsigned short* h = H1 + (size_t)b * HID;
        const float* w = fcW + (size_t)o * HID;
        float acc = fcb[o];
        for (int k = 0; k < HID; ++k) acc += bf2f(h[k]) * w[k];
        out[(size_t)b * (TSTEPS * NOUT) + t * NOUT + o] = acc;
        s[o] = acc;
    }
    __syncthreads();
    if (o == 0) {
        int best = 0; float bv = s[0];
        for (int i = 1; i < NACT; ++i) if (s[i] > bv) { bv = s[i]; best = i; }
        act[b] = best;
    }
}

// in-place: log_softmax over first 16 channels per (b,t); softmax over time for ch16
__global__ __launch_bounds__(64)
void post_kernel(float* __restrict__ out) {
    int b = blockIdx.x, t = threadIdx.x;
    float* row = out + (size_t)b * TSTEPS * NOUT + (size_t)t * NOUT;
    float m = row[0];
    for (int i = 1; i < NACT; ++i) m = fmaxf(m, row[i]);
    float s = 0.0f;
    for (int i = 0; i < NACT; ++i) s += expf(row[i] - m);
    float lse = m + logf(s);
    float d = row[16];
    for (int i = 0; i < NACT; ++i) row[i] -= lse;

    __shared__ float sm[64], ss[64];
    sm[t] = d; __syncthreads();
    for (int off = 32; off > 0; off >>= 1) {
        if (t < off) sm[t] = fmaxf(sm[t], sm[t + off]);
        __syncthreads();
    }
    float e = expf(d - sm[0]);
    ss[t] = e; __syncthreads();
    for (int off = 32; off > 0; off >>= 1) {
        if (t < off) ss[t] += ss[t + off];
        __syncthreads();
    }
    row[16] = e / ss[0];
}

// ---------------------------------------------------------------------------
extern "C" void kernel_launch(void* const* d_in, const int* in_sizes, int n_in,
                              void* d_out, int out_size, void* d_ws, size_t ws_size,
                              hipStream_t stream) {
    (void)in_sizes; (void)n_in; (void)out_size; (void)ws_size;
    const float* hidden = (const float*)d_in[1];   // (L,B,H)
    const float* cell   = (const float*)d_in[2];   // (L,B,H)
    const float* emb    = (const float*)d_in[3];   // (V,H-1)
    const float* W_ih   = (const float*)d_in[4];   // (L,4H,H)
    const float* W_hh   = (const float*)d_in[5];   // (L,4H,H)
    const float* b_ih   = (const float*)d_in[6];   // (L,4H)
    const float* b_hh   = (const float*)d_in[7];   // (L,4H)
    const float* fc_W   = (const float*)d_in[8];   // (O,H)
    const float* fc_b   = (const float*)d_in[9];   // (O,)
    float* out = (float*)d_out;

    // workspace carve-up
    char* ws = (char*)d_ws;
    unsigned short* WihB = (unsigned short*)ws; ws += (size_t)NLAY * NGATE * HID * 2;
    unsigned short* WhhB = (unsigned short*)ws; ws += (size_t)NLAY * NGATE * HID * 2;
    unsigned short* Xb   = (unsigned short*)ws; ws += (size_t)BATCH * HID * 2;
    unsigned short* Hb   = (unsigned short*)ws; ws += (size_t)NLAY * BATCH * HID * 2;
    float*          Cb   = (float*)ws;          ws += (size_t)NLAY * BATCH * HID * 4;
    float*          Gb   = (float*)ws;          ws += (size_t)BATCH * NGATE * 4;
    int*            act  = (int*)ws;            ws += (size_t)BATCH * 4;

    const int nW = NLAY * NGATE * HID;         // 2,097,152
    const int nS = NLAY * BATCH * HID;         // 1,048,576
    cvt_bf16_kernel<<<(nW + 255) / 256, 256, 0, stream>>>(W_ih, WihB, nW);
    cvt_bf16_kernel<<<(nW + 255) / 256, 256, 0, stream>>>(W_hh, WhhB, nW);
    cvt_bf16_kernel<<<(nS + 255) / 256, 256, 0, stream>>>(hidden, Hb, nS);
    hipMemcpyAsync(Cb, cell, (size_t)nS * sizeof(float), hipMemcpyDeviceToDevice, stream);

    const dim3 gGrid(BATCH / 64, NGATE / 128);         // 16 x 16 workgroups
    const int  eBlocks = (BATCH * HID) / 256;          // 2048
    const int  cBlocks = (BATCH * HID) / 256;          // 2048

    for (int t = 0; t < TSTEPS; ++t) {
        embed_kernel<<<eBlocks, 256, 0, stream>>>(emb, act, t == 0 ? 1 : 0, Xb);
        // layer 0
        gates_kernel<<<gGrid, 256, 0, stream>>>(Xb, Hb, WihB, WhhB, Gb);
        cell_kernel<<<cBlocks, 256, 0, stream>>>(Gb, b_ih, b_hh, Cb, Hb);
        // layer 1 (input = layer-0 h just produced)
        gates_kernel<<<gGrid, 256, 0, stream>>>(Hb, Hb + (size_t)BATCH * HID,
                                                WihB + (size_t)NGATE * HID,
                                                WhhB + (size_t)NGATE * HID, Gb);
        cell_kernel<<<cBlocks, 256, 0, stream>>>(Gb, b_ih + NGATE, b_hh + NGATE,
                                                 Cb + (size_t)BATCH * HID,
                                                 Hb + (size_t)BATCH * HID);
        // FC + argmax -> next action
        fc_kernel<<<BATCH, 32, 0, stream>>>(Hb + (size_t)BATCH * HID, fc_W, fc_b,
                                            out, act, t);
    }
    post_kernel<<<BATCH, 64, 0, stream>>>(out);
}
